// NaiveAttention_29764123361373
// MI455X (gfx1250) — compile-verified, matrix operand order checked by reading, not run on hardware
//
#include <hip/hip_runtime.h>

// ---------------------------------------------------------------------------
// CDNA5 (gfx1250) fused multi-head attention, bf16 WMMA path.
//   B=2, S=2048, D=1024, H=16, Dh=64.  All matmuls via v_wmma_f32_16x16x32_bf16.
//   GEMM tiles staged with GLOBAL_LOAD_ASYNC_TO_LDS_B128 (double-buffered).
// ---------------------------------------------------------------------------

typedef __attribute__((ext_vector_type(16))) __bf16 v16bf;
typedef __attribute__((ext_vector_type(8)))  __bf16 v8bf;
typedef __attribute__((ext_vector_type(4)))  __bf16 v4bf;
typedef __attribute__((ext_vector_type(8)))  float  v8f;
typedef int v4i_async __attribute__((vector_size(16)));   // <4 x i32> payload type

#define LOG2E_F 1.4426950408889634f
#define AS3 __attribute__((address_space(3)))

#if defined(__has_builtin)
#  if __has_builtin(__builtin_amdgcn_global_load_async_to_lds_b128)
#    define HAVE_ASYNC_LDS 1
#  endif
#endif
#ifndef HAVE_ASYNC_LDS
#  define HAVE_ASYNC_LDS 0
#endif

// Async 16-byte global -> LDS copy (ASYNCcnt-tracked); sync fallback.
static __device__ __forceinline__ void async_copy16(const __bf16* g, __bf16* l) {
#if HAVE_ASYNC_LDS
  __builtin_amdgcn_global_load_async_to_lds_b128(
      (v4i_async*)(g), (AS3 v4i_async*)(l), 0, 0);
#else
  *(v8bf*)l = *(const v8bf*)g;
#endif
}

static __device__ __forceinline__ void wait_async_all() {
#if HAVE_ASYNC_LDS
#  if __has_builtin(__builtin_amdgcn_s_wait_asynccnt)
  __builtin_amdgcn_s_wait_asynccnt(0);
#  else
  asm volatile("s_wait_asynccnt 0" ::: "memory");
#  endif
#endif
}

// Build a 16-element bf16 A/B fragment from two contiguous 8-element runs.
// Per ISA layout (16-bit A 16x32 / B 32x16): elements 0..7 live at column
// offset half*8, elements 8..15 at column offset 16+half*8 (p and p+16).
static __device__ __forceinline__ v16bf load_frag(const __bf16* p) {
  v8bf lo = *(const v8bf*)(p);
  v8bf hi = *(const v8bf*)(p + 16);
  return __builtin_shufflevector(lo, hi, 0,1,2,3,4,5,6,7,8,9,10,11,12,13,14,15);
}

static __device__ __forceinline__ v8f wmma_bf16(v16bf a, v16bf b, v8f c) {
  // (neg_a, A, neg_b, B, c_mod, C, reuse_a, reuse_b)
  return __builtin_amdgcn_wmma_f32_16x16x32_bf16(false, a, false, b,
                                                 (short)0, c, false, false);
}

// ---------------------------------------------------------------------------
// fp32 -> bf16 conversion (4 elems / thread, fully vectorized)
// ---------------------------------------------------------------------------
__global__ __launch_bounds__(256) void mha_cvt_bf16(const float* __restrict__ s,
                                                    __bf16* __restrict__ d, int n) {
  int i = (blockIdx.x * 256 + threadIdx.x) * 4;
  if (i >= n) return;
  float4 v = *(const float4*)(s + i);
  v4bf o;
  o[0] = (__bf16)v.x; o[1] = (__bf16)v.y; o[2] = (__bf16)v.z; o[3] = (__bf16)v.w;
  *(v4bf*)(d + i) = o;
}

// ---------------------------------------------------------------------------
// fp32 [K=1024, N=1024] -> bf16 TRANSPOSED [N, K] (LDS-tiled, 32x32 tiles).
// Transposing the weights once makes every GEMM LDS stage a contiguous copy.
// ---------------------------------------------------------------------------
__global__ __launch_bounds__(256) void mha_cvt_transpose_bf16(
    const float* __restrict__ W, __bf16* __restrict__ Wt) {
  __shared__ float tile[32][33];               // +1 pad: no bank conflicts
  const int bn = blockIdx.x * 32;              // N tile origin
  const int bk = blockIdx.y * 32;              // K tile origin
  const int tx = threadIdx.x & 31;
  const int ty = threadIdx.x >> 5;             // 0..7
  #pragma unroll
  for (int i = 0; i < 32; i += 8)
    tile[ty + i][tx] = W[(size_t)(bk + ty + i) * 1024 + bn + tx];
  __syncthreads();
  #pragma unroll
  for (int i = 0; i < 32; i += 8)
    Wt[(size_t)(bn + ty + i) * 1024 + bk + tx] = (__bf16)tile[tx][ty + i];
}

// ---------------------------------------------------------------------------
// Tiled bf16 GEMM:  C[M,N] = A[M,K] * Wt[N,K]^T + bias,  M=4096, N=K=1024.
// A row-major [m][k]; W pre-transposed [n][k] -> both tiles are contiguous
// async B128 copies into double-buffered LDS.
// Block tile 128x128, 8 waves (2 in M x 4 in N), wave tile 64x32, K-step 32.
// MODE 0: bf16 row-major out.  MODE 1: bf16 out transposed per head
// ([B,H,Dh,S] for V).  MODE 2: fp32 row-major out (final projection).
// ---------------------------------------------------------------------------
template<int MODE>
__global__ __launch_bounds__(256) void mha_gemm_bf16(const __bf16* __restrict__ A,
                                                     const __bf16* __restrict__ Wt,
                                                     const float*  __restrict__ bias,
                                                     void* __restrict__ outp) {
  constexpr int Ndim = 1024, Kdim = 1024;
  __shared__ __bf16 sA[2][128 * 32];   // [m][k]
  __shared__ __bf16 sB[2][128 * 32];   // [n][k]

  const int tid  = threadIdx.x;
  const int lane = tid & 31;
  const int wave = tid >> 5;
  const int half = lane >> 4;
  const int ln   = lane & 15;
  const int m0   = blockIdx.y * 128;
  const int n0   = blockIdx.x * 128;
  const int mOff = (wave >> 2) * 64;   // 2 waves in M
  const int nOff = (wave & 3) * 32;    // 4 waves in N

  auto stage = [&](int buf, int kk) {
    #pragma unroll
    for (int i = 0; i < 2; ++i) {
      int c = i * 256 + tid;
      int row = c >> 2, cc = (c & 3) * 8;
      async_copy16(&A[(size_t)(m0 + row) * Kdim + kk + cc],
                   &sA[buf][row * 32 + cc]);
      async_copy16(&Wt[(size_t)(n0 + row) * Kdim + kk + cc],
                   &sB[buf][row * 32 + cc]);
    }
  };

  v8f acc[4][2];
  const v8f vzero = {0.f,0.f,0.f,0.f,0.f,0.f,0.f,0.f};
  #pragma unroll
  for (int mt = 0; mt < 4; ++mt)
    #pragma unroll
    for (int nt = 0; nt < 2; ++nt) acc[mt][nt] = vzero;

  stage(0, 0);
  for (int kk = 0; kk < Kdim; kk += 32) {
    const int cur = (kk >> 5) & 1;
    wait_async_all();          // this wave's staged writes are in LDS
    __syncthreads();           // ...and everyone else's too
    if (kk + 32 < Kdim) stage(cur ^ 1, kk + 32);   // stream next tile

    v16bf af[4], bf[2];
    #pragma unroll
    for (int mt = 0; mt < 4; ++mt)
      af[mt] = load_frag(&sA[cur][(mOff + mt * 16 + ln) * 32 + half * 8]);
    #pragma unroll
    for (int nt = 0; nt < 2; ++nt)
      bf[nt] = load_frag(&sB[cur][(nOff + nt * 16 + ln) * 32 + half * 8]);
    #pragma unroll
    for (int mt = 0; mt < 4; ++mt)
      #pragma unroll
      for (int nt = 0; nt < 2; ++nt)
        acc[mt][nt] = wmma_bf16(af[mt], bf[nt], acc[mt][nt]);
    __syncthreads();           // done reading `cur` before it is re-staged
  }

  // ---- epilogue: D layout lane(l): col = l%16, row(vgpr r) = r + 8*(l/16) ----
  #pragma unroll
  for (int nt = 0; nt < 2; ++nt) {
    int col  = n0 + nOff + nt * 16 + ln;
    float bv = bias[col];
    #pragma unroll
    for (int mt = 0; mt < 4; ++mt) {
      #pragma unroll
      for (int r = 0; r < 8; ++r) {
        int row = m0 + mOff + mt * 16 + r + 8 * half;
        float v = acc[mt][nt][r] + bv;
        if constexpr (MODE == 0) {
          ((__bf16*)outp)[(size_t)row * Ndim + col] = (__bf16)v;
        } else if constexpr (MODE == 1) {
          // V transposed: [B,H,Dh,S];  row = b*2048+s, col = h*64+dh
          int b = row >> 11, s = row & 2047;
          int h = col >> 6,  dh = col & 63;
          ((__bf16*)outp)[(((size_t)(b * 16 + h) * 64 + dh) << 11) + s] = (__bf16)v;
        } else {
          ((float*)outp)[(size_t)row * Ndim + col] = v;
        }
      }
    }
  }
}

// ---------------------------------------------------------------------------
// Flash-attention: one 16-query tile per wave, online softmax, all matmuls
// via WMMA.  Q/K stored bf16 [B,S,H*Dh]; V stored bf16 transposed [B,H,Dh,S].
// Running max kept in the RAW score domain; SCALE folded into exp2 argument.
// ---------------------------------------------------------------------------
__global__ __launch_bounds__(256) void mha_flash_attn(const __bf16* __restrict__ qb,
                                                      const __bf16* __restrict__ kb,
                                                      const __bf16* __restrict__ vt,
                                                      __bf16* __restrict__ ob) {
  constexpr float CEXP = 0.125f * LOG2E_F;   // SCALE * log2(e)
  __shared__ __bf16 ldsP[8][16 * 64];        // per-wave P tile [16 q][64 keys]

  const int tid  = threadIdx.x;
  const int lane = tid & 31;
  const int wave = tid >> 5;
  const int half = lane >> 4;
  const int ln   = lane & 15;

  const int t  = blockIdx.x * 8 + wave;   // 4096 query tiles total
  const int qt = t & 127;                 // S/16 = 128 tiles per (b,h)
  const int h  = (t >> 7) & 15;
  const int b  = t >> 11;
  const int q0 = qt * 16;
  __bf16* P = ldsP[wave];

  // Persistent Q A-fragments: Dh=64 -> 2 K-steps of 32.
  v16bf qf[2];
  {
    const __bf16* qrow = qb + (size_t)(b * 2048 + q0 + ln) * 1024 + h * 64;
    qf[0] = load_frag(qrow + half * 8);
    qf[1] = load_frag(qrow + 32 + half * 8);
  }

  const v8f vzero = {0.f,0.f,0.f,0.f,0.f,0.f,0.f,0.f};
  v8f oacc[4];
  #pragma unroll
  for (int nto = 0; nto < 4; ++nto) oacc[nto] = vzero;
  float mrow[8], lrow[8];
  #pragma unroll
  for (int r = 0; r < 8; ++r) { mrow[r] = -3.0e38f; lrow[r] = 0.f; }

  const __bf16* kbase = kb + (size_t)b * 2048 * 1024 + h * 64;
  const __bf16* vbase = vt + ((size_t)(b * 16 + h) * 64) * 2048;

  for (int k0 = 0; k0 < 2048; k0 += 64) {
    // ---- S tile = Q * K^T : 16 q x 64 keys ----
    v8f sacc[4];
    #pragma unroll
    for (int nt = 0; nt < 4; ++nt) sacc[nt] = vzero;
    #pragma unroll
    for (int ks = 0; ks < 2; ++ks) {
      #pragma unroll
      for (int nt = 0; nt < 4; ++nt) {
        // K is row-major [key][Dh] == exactly the B-fragment layout for QK^T
        v16bf kf = load_frag(kbase + (size_t)(k0 + nt * 16 + ln) * 1024
                             + ks * 32 + half * 8);
        sacc[nt] = wmma_bf16(qf[ks], kf, sacc[nt]);
      }
    }

    // ---- online softmax (rows live in 16-lane groups, raw-score domain) ----
    #pragma unroll
    for (int r = 0; r < 8; ++r) {
      float mx = fmaxf(fmaxf(sacc[0][r], sacc[1][r]),
                       fmaxf(sacc[2][r], sacc[3][r]));
      #pragma unroll
      for (int mask = 1; mask <= 8; mask <<= 1)
        mx = fmaxf(mx, __shfl_xor(mx, mask, 32));
      float mnew  = fmaxf(mrow[r], mx);
      float alpha = exp2f((mrow[r] - mnew) * CEXP);
      mrow[r] = mnew;
      float rs = 0.f;
      #pragma unroll
      for (int nt = 0; nt < 4; ++nt) {
        float p = exp2f((sacc[nt][r] - mnew) * CEXP);
        sacc[nt][r] = p;
        rs += p;
      }
      #pragma unroll
      for (int mask = 1; mask <= 8; mask <<= 1)
        rs += __shfl_xor(rs, mask, 32);
      lrow[r] = lrow[r] * alpha + rs;
      #pragma unroll
      for (int nto = 0; nto < 4; ++nto) oacc[nto][r] *= alpha;
      // stash P row as bf16 into wave-private LDS (in-order per wave)
      #pragma unroll
      for (int nt = 0; nt < 4; ++nt)
        P[(r + 8 * half) * 64 + nt * 16 + ln] = (__bf16)sacc[nt][r];
    }

    // ---- O += P * V : A-frags of P from LDS, B-frags of V^T from global ----
    #pragma unroll
    for (int ks = 0; ks < 2; ++ks) {
      v16bf pf = load_frag(&P[ln * 64 + ks * 32 + half * 8]);
      #pragma unroll
      for (int nto = 0; nto < 4; ++nto) {
        v16bf vf = load_frag(vbase + (size_t)(nto * 16 + ln) * 2048
                             + k0 + ks * 32 + half * 8);
        oacc[nto] = wmma_bf16(pf, vf, oacc[nto]);
      }
    }
  }

  // ---- normalize and write attn output bf16 [B,S,H*Dh] ----
  #pragma unroll
  for (int nto = 0; nto < 4; ++nto) {
    int dh = nto * 16 + ln;
    #pragma unroll
    for (int r = 0; r < 8; ++r) {
      int row = q0 + r + 8 * half;
      float v = oacc[nto][r] / lrow[r];
      ob[(size_t)(b * 2048 + row) * 1024 + h * 64 + dh] = (__bf16)v;
    }
  }
}

// ---------------------------------------------------------------------------
// Host launch
// ---------------------------------------------------------------------------
extern "C" void kernel_launch(void* const* d_in, const int* in_sizes, int n_in,
                              void* d_out, int out_size, void* d_ws, size_t ws_size,
                              hipStream_t stream) {
  (void)in_sizes; (void)n_in; (void)out_size; (void)ws_size;
  const float* x  = (const float*)d_in[0];
  const float* Wq = (const float*)d_in[1];
  const float* bq = (const float*)d_in[2];
  const float* Wk = (const float*)d_in[3];
  const float* bk = (const float*)d_in[4];
  const float* Wv = (const float*)d_in[5];
  const float* bv = (const float*)d_in[6];
  const float* Wo = (const float*)d_in[7];
  const float* bo = (const float*)d_in[8];

  const size_t MELEM = 4096 * 1024;   // x / Q / K / V / attn
  const size_t WELEM = 1024 * 1024;   // each weight
  __bf16* xb   = (__bf16*)d_ws;
  __bf16* wqt  = xb + MELEM;          // transposed bf16 weights [N,K]
  __bf16* wkt  = wqt + WELEM;
  __bf16* wvt  = wkt + WELEM;
  __bf16* wot  = wvt + WELEM;
  __bf16* qbuf = wot + WELEM;
  __bf16* kbuf = qbuf + MELEM;
  __bf16* vtb  = kbuf + MELEM;
  __bf16* atb  = vtb + MELEM;

  // fp32 -> bf16 staging (x plain; weights transposed to [N,K])
  mha_cvt_bf16<<<4096, 256, 0, stream>>>(x, xb, (int)MELEM);
  dim3 gt(32, 32);
  mha_cvt_transpose_bf16<<<gt, 256, 0, stream>>>(Wq, wqt);
  mha_cvt_transpose_bf16<<<gt, 256, 0, stream>>>(Wk, wkt);
  mha_cvt_transpose_bf16<<<gt, 256, 0, stream>>>(Wv, wvt);
  mha_cvt_transpose_bf16<<<gt, 256, 0, stream>>>(Wo, wot);

  dim3 g(8, 32), blk(256);
  mha_gemm_bf16<0><<<g, blk, 0, stream>>>(xb, wqt, bq, qbuf);  // Q [B,S,D] bf16
  mha_gemm_bf16<0><<<g, blk, 0, stream>>>(xb, wkt, bk, kbuf);  // K [B,S,D] bf16
  mha_gemm_bf16<1><<<g, blk, 0, stream>>>(xb, wvt, bv, vtb);   // V^T [B,H,Dh,S] bf16

  mha_flash_attn<<<512, 256, 0, stream>>>(qbuf, kbuf, vtb, atb);

  mha_gemm_bf16<2><<<g, blk, 0, stream>>>(atb, wot, bo, d_out); // fp32 out
}
